// QuantizedConv2d_1864015806713
// MI455X (gfx1250) — 8-bit matrix operand order repaired and checked by reading, compiled, not run on hardware
//
#include <hip/hip_runtime.h>
#include <stdint.h>


typedef int v8i __attribute__((ext_vector_type(8)));

#define OUT_ZP      5
#define SCALE_RATIO 0.5f
#define Bn 32
#define Cn 128
#define Hn 112
#define Wn 112

#define NROWS     18
#define ROWSTRIDE 144

template <int KOFF>
__device__ __forceinline__ v8i build_bfrag(const int w9[9], int n) {
  v8i bf;
#pragma unroll
  for (int v = 0; v < 8; ++v) {
    unsigned dw = 0;
#pragma unroll
    for (int byi = 0; byi < 4; ++byi) {
      const int k = (KOFF / 2) + (v >> 1) * 16 + (v & 1) * 4 + byi;
      int val = 0;
      if (k < 54) {
        const int dy = k / 18;
        const int j  = k - dy * 18;
        const int d  = j - n;
        val = (d == 0) ? w9[dy * 3 + 0]
            : (d == 1) ? w9[dy * 3 + 1]
            : (d == 2) ? w9[dy * 3 + 2] : 0;
      }
      dw |= (unsigned)(val & 0xFF) << (8 * byi);
    }
    bf[v] = (int)dw;
  }
  return bf;
}

__global__ __launch_bounds__(32)
void dwconv3x3_q8_wmma(const int* __restrict__ x,
                       const int* __restrict__ weight,
                       const int* __restrict__ bias,
                       const float* __restrict__ wscale,
                       int8_t* __restrict__ out)
{
  __shared__ unsigned char lds[NROWS * ROWSTRIDE];

  const int lane = threadIdx.x;
  const int gid  = blockIdx.x;
  const int ty   = gid % 7;
  const int bc   = gid / 7;
  const int c    = bc % Cn;
  const int b    = bc / Cn;
  const int y0   = ty * 16;

  int w9[9];
#pragma unroll
  for (int i = 0; i < 9; ++i) w9[i] = weight[c * 9 + i];
  const int   bias_c  = bias[c];
  const float scale_c = SCALE_RATIO * wscale[c];

  const int  n  = lane & 15;
  const bool hi = lane >= 16;

  const v8i bfrag = hi ? build_bfrag<16>(w9, n) : build_bfrag<0>(w9, n);

  const size_t plane = (size_t)(b * Cn + c) * Hn;
#pragma unroll 1
  for (int r = 0; r < NROWS; ++r) {
    const int gy = y0 - 1 + r;
    const int rb = r * ROWSTRIDE;
    if (lane < 28) {
      unsigned packed = 0xFDFDFDFDu;
      if (gy >= 0 && gy < Hn) {
        const int4 v4 = *(const int4*)(x + (plane + gy) * Wn + lane * 4);
        packed = (unsigned)(v4.x & 0xFF)        |
                 ((unsigned)(v4.y & 0xFF) << 8) |
                 ((unsigned)(v4.z & 0xFF) << 16)|
                 ((unsigned)v4.w << 24);
      }
      *(unsigned*)(&lds[rb + 16 + lane * 4]) = packed;
    } else if (lane == 28) {
      lds[rb + 15]  = 0xFDu;
    } else if (lane == 29) {
      lds[rb + 128] = 0xFDu;
    }
  }
  __syncthreads();

  const int m = lane & 15;

#pragma unroll 1
  for (int tx = 0; tx < 7; ++tx) {
    const int x0 = tx * 16;

    unsigned w[3][5];
#pragma unroll
    for (int dy = 0; dy < 3; ++dy) {
      const int base = (m + dy) * ROWSTRIDE + 16 + x0;
      const unsigned d0 = *(const unsigned*)(&lds[base - 4]);
      const uint4    dm = *(const uint4*)(&lds[base]);
      const unsigned d5 = *(const unsigned*)(&lds[base + 16]);
      w[dy][0] = (d0   >> 24) | (dm.x << 8);
      w[dy][1] = (dm.x >> 24) | (dm.y << 8);
      w[dy][2] = (dm.y >> 24) | (dm.z << 8);
      w[dy][3] = (dm.z >> 24) | (dm.w << 8);
      w[dy][4] = (dm.w >> 24) | (d5   << 8);
    }

    v8i afrag;
    if (!hi) {
      afrag[0] = (int)w[0][0];
      afrag[1] = (int)w[0][1];
      afrag[2] = (int)((w[0][4] & 0xFFFFu) | (w[1][0] << 16));
      afrag[3] = (int)((w[1][0] >> 16) | (w[1][1] << 16));
      afrag[4] = (int)((w[1][3] >> 16) | (w[1][4] << 16));
      afrag[5] = (int)w[2][0];
      afrag[6] = (int)w[2][3];
      afrag[7] = (int)(w[2][4] & 0xFFFFu);
    } else {
      afrag[0] = (int)w[0][2];
      afrag[1] = (int)w[0][3];
      afrag[2] = (int)((w[1][1] >> 16) | (w[1][2] << 16));
      afrag[3] = (int)((w[1][2] >> 16) | (w[1][3] << 16));
      afrag[4] = (int)w[2][1];
      afrag[5] = (int)w[2][2];
      afrag[6] = 0;
      afrag[7] = 0;
    }

    v8i acc = {};
    acc = __builtin_amdgcn_wmma_i32_16x16x64_iu8(
         true, afrag,  true, bfrag, acc,
         false,  false);

    int8_t* orow = out + (plane + y0) * (size_t)Wn + x0 + n;
#pragma unroll
    for (int g = 0; g < 8; ++g) {
      const int mrow = g + (hi ? 8 : 0);
      float f = (float)(acc[g] + bias_c) * scale_c + (float)OUT_ZP;
      f = rintf(f);
      f = fminf(fmaxf(f, -128.0f), 127.0f);
      orow[(size_t)mrow * Wn] = (int8_t)(int)f;
    }
  }
}

extern "C" void kernel_launch(void* const* d_in, const int* in_sizes, int n_in,
                              void* d_out, int out_size, void* d_ws, size_t ws_size,
                              hipStream_t stream) {
  const int*   x  = (const int*)d_in[0];
  const int*   wt = (const int*)d_in[1];
  const int*   bs = (const int*)d_in[2];
  const float* ws = (const float*)d_in[3];
  int8_t* out = (int8_t*)d_out;

  dim3 grid(Bn * Cn * 7);
  dim3 block(32);
  hipLaunchKernelGGL(dwconv3x3_q8_wmma, grid, block, 0, stream,
                     x, wt, bs, ws, out);
}
